// ContinuousFilterConvolution_56092272886168
// MI455X (gfx1250) — compile-verified, matrix operand order checked by reading, not run on hardware
//
#include <hip/hip_runtime.h>

#define DH 128
#define NB 16

typedef __attribute__((ext_vector_type(16))) _Float16 v16h;
typedef __attribute__((ext_vector_type(8)))  _Float16 v8h;
typedef __attribute__((ext_vector_type(8)))  float    v8f;

union V16U { v16h v; v8h h[2]; };

__global__ void cfconv_zero_kernel(float* __restrict__ out, int n) {
    int i = blockIdx.x * blockDim.x + threadIdx.x;
    if (i < n) out[i] = 0.0f;
}

// Block = 128 threads = 4 waves; each wave processes one 16-edge tile per loop step.
__global__ void __launch_bounds__(128)
cfconv_wmma_kernel(const float* __restrict__ node_feats,
                   const float* __restrict__ coords,
                   const int*   __restrict__ src,
                   const int*   __restrict__ dest,
                   const float* __restrict__ W1,
                   const float* __restrict__ W2,
                   float*       __restrict__ out,
                   int nTiles, int E)
{
    // f16 staged weights: W2 transposed [n][k], W1 transposed + K-padded to 32 [n][32]
    __shared__ __attribute__((aligned(16))) _Float16 sW2T[DH * DH];
    __shared__ __attribute__((aligned(16))) _Float16 sW1T[DH * 32];
    __shared__ __attribute__((aligned(16))) _Float16 sM[4][16 * DH]; // per-wave m tile
    __shared__ int sSrc[4][16];
    __shared__ int sDst[4][16];

    const int tid  = threadIdx.x;
    const int wave = tid >> 5;
    const int lane = tid & 31;
    const int l16  = lane & 15;
    const int hi   = lane >> 4;          // half-wave select (K-group / M-group)

    // ---- stage weights into LDS (once per block) ----
    for (int i = tid; i < DH * DH; i += blockDim.x) {
        int k = i >> 7, n = i & 127;
        sW2T[n * DH + k] = (_Float16)W2[i];          // W2 is [k][n] row-major
    }
    for (int i = tid; i < DH * 32; i += blockDim.x) {
        int n = i >> 5, kk = i & 31;
        sW1T[i] = (kk < NB) ? (_Float16)W1[kk * DH + n] : (_Float16)0.0f;
    }
    __syncthreads();

    const float width = 4.5f / 15.0f;                // (D_MAX-D_MIN)/(NB-1)
    const float coeff = -0.5f / (width * width);
    const int   tileStride = gridDim.x * 4;
    _Float16* mrow = &sM[wave][0];

    for (int tile = blockIdx.x * 4 + wave; tile < nTiles; tile += tileStride) {
        const int ebase = tile * 16;

        // ---- per-edge distance + RBF (lane L: edge l16, bases 8*hi..8*hi+7) ----
        int e = ebase + l16;
        bool live = (e < E);
        int s = 0, t = 0;
        if (live) { s = src[e]; t = dest[e]; }
        // hide next iteration's index-load latency (global_prefetch_b8)
        {
            int en = e + tileStride * 16;
            if (en < E) {
                __builtin_prefetch(src  + en, 0, 0);
                __builtin_prefetch(dest + en, 0, 0);
            }
        }
        float dx = coords[3 * s + 0] - coords[3 * t + 0];
        float dy = coords[3 * s + 1] - coords[3 * t + 1];
        float dz = coords[3 * s + 2] - coords[3 * t + 2];
        float d  = sqrtf(dx * dx + dy * dy + dz * dz);
        if (hi == 0) { sSrc[wave][l16] = s; sDst[wave][l16] = t; }

        // A fragment for GEMM1 lands directly in WMMA layout (K 16..31 padded zero)
        V16U aR;
        const float x0 = d - (float)(8 * hi) * width;
        #pragma unroll
        for (int j = 0; j < 8; ++j) {
            float x = x0 - (float)j * width;
            float r = live ? __expf(coeff * x * x) : 0.0f;
            aR.v[j]     = (_Float16)r;
            aR.v[j + 8] = (_Float16)0.0f;
        }

        // ---- GEMM1: m = relu(rbf @ W1)   (8 x v_wmma_f32_16x16x32_f16) ----
        #pragma unroll
        for (int nt = 0; nt < 8; ++nt) {
            int col = nt * 16 + l16;
            V16U b;
            b.h[0] = *(const v8h*)&sW1T[col * 32 + 8 * hi];
            b.h[1] = *(const v8h*)&sW1T[col * 32 + 16 + 8 * hi];
            v8f c = {};
            c = __builtin_amdgcn_wmma_f32_16x16x32_f16(false, aR.v, false, b.v,
                                                       (short)0, c, false, false);
            // D element r -> (M = r + 8*hi, N = col); relu + f16 into wave-private tile
            _Float16* mp = mrow + (8 * hi) * DH + col;
            #pragma unroll
            for (int r = 0; r < 8; ++r) {
                mp[r * DH] = (_Float16)fmaxf(c[r], 0.0f);
            }
        }

        // ---- GEMM2: y = relu(m @ W2)   (4 K-chunks x 8 N-tiles = 32 WMMAs) ----
        v8f acc[8] = {};
        #pragma unroll
        for (int kt = 0; kt < 4; ++kt) {
            int kb = kt * 32 + 8 * hi;
            V16U a;
            a.h[0] = *(const v8h*)&mrow[l16 * DH + kb];       // ds_load_b128
            a.h[1] = *(const v8h*)&mrow[l16 * DH + kb + 16];
            #pragma unroll
            for (int nt = 0; nt < 8; ++nt) {
                int col = nt * 16 + l16;
                V16U b;
                b.h[0] = *(const v8h*)&sW2T[col * DH + kb];
                b.h[1] = *(const v8h*)&sW2T[col * DH + kb + 16];
                acc[nt] = __builtin_amdgcn_wmma_f32_16x16x32_f16(false, a.v, false, b.v,
                                                                 (short)0, acc[nt],
                                                                 false, false);
            }
        }

        // ---- gather node_feats[src] * relu(y), scatter-add into out[dest] ----
        // Row-outer: one src/dest read + one base address per edge row; the 8
        // feature columns become compile-time immediate offsets (nt*64 bytes).
        #pragma unroll
        for (int r = 0; r < 8; ++r) {
            int m  = r + 8 * hi;
            int em = ebase + m;
            if (em < E) {
                int sm = sSrc[wave][m];
                int dm = sDst[wave][m];
                const float* __restrict__ nf = node_feats + sm * DH + l16;
                float*       __restrict__ op = out        + dm * DH + l16;
                #pragma unroll
                for (int nt = 0; nt < 8; ++nt) {
                    float v = fmaxf(acc[nt][r], 0.0f) * nf[nt * 16];
                    atomicAdd(&op[nt * 16], v);
                }
            }
        }
    }
}

extern "C" void kernel_launch(void* const* d_in, const int* in_sizes, int n_in,
                              void* d_out, int out_size, void* d_ws, size_t ws_size,
                              hipStream_t stream) {
    const float* node_feats = (const float*)d_in[0];
    const float* coords     = (const float*)d_in[1];
    const int*   src        = (const int*)d_in[2];
    const int*   dest       = (const int*)d_in[3];
    const float* W1         = (const float*)d_in[4];
    const float* W2         = (const float*)d_in[5];
    float*       out        = (float*)d_out;

    const int E      = in_sizes[2];
    const int nTiles = (E + 15) / 16;

    // out is poisoned by the harness; segment-sum needs zeros.
    cfconv_zero_kernel<<<(out_size + 255) / 256, 256, 0, stream>>>(out, out_size);

    int blocks = (nTiles + 3) / 4;
    if (blocks > 4096) blocks = 4096;   // persistent grid-stride: amortize weight staging
    if (blocks < 1) blocks = 1;
    cfconv_wmma_kernel<<<blocks, 128, 0, stream>>>(node_feats, coords, src, dest,
                                                   W1, W2, out, nTiles, E);
}
